// MultiHeadAttention_10479720202276
// MI455X (gfx1250) — matrix operand order repaired and checked by reading, not compiled, not run on hardware
//
#include <hip/hip_runtime.h>

// Fused per-token multi-head-over-heads attention for MI455X (gfx1250, wave32).
// - Q/K/V/O projections on v_wmma_f32_16x16x32_f16 (f32 accumulate)
// - weights pre-converted to f16 once (prep kernel -> d_ws), then staged into
//   LDS per tile with global_load_async_to_lds_b128 (ASYNCcnt path)
// - tiny per-token 32x32 softmax over the heads axis on VALU (v_exp_f32)
// Fully fused: x read once, out written once (~64MB HBM @ 23.3TB/s ~ 2.7us).

typedef _Float16 v16h __attribute__((ext_vector_type(16)));
typedef _Float16 v8h  __attribute__((ext_vector_type(8)));
typedef _Float16 v4h  __attribute__((ext_vector_type(4)));
typedef float    v8f  __attribute__((ext_vector_type(8)));

#define EMBED 128
#define HEADS 32
#define HDIM  4
#define NTOK  (32 * 2048)
#define TOK_PER_BLOCK 128
#define NTILES (NTOK / TOK_PER_BLOCK)   /* 512 */
#define THREADS 256                     /* 8 wave32s */
#define WELEMS (EMBED * EMBED)          /* 16384 */

struct Smem {
  _Float16 wstage[WELEMS];                 // 32 KB: current weight (f16, [out][in])
  _Float16 qkv[3][TOK_PER_BLOCK * EMBED];  // 96 KB; qkv[0] reused as attended buf
};                                          // 128 KB total (< 160 KB CU-mode limit)

static __device__ __forceinline__ v16h cat8(v8h lo, v8h hi) {
  return __builtin_shufflevector(lo, hi, 0,1,2,3,4,5,6,7,8,9,10,11,12,13,14,15);
}

// A operand (16x32 f16): lane l<16 = row M=l, halves 0..7 = K kbase..kbase+7,
// halves 8..15 = K kbase+16..kbase+23; lanes 16..31 same rows, K offset +8.
static __device__ __forceinline__ v16h make_afrag_f32(const float* __restrict__ row, int kbase) {
  float4 f0 = *(const float4*)(row + kbase);
  float4 f1 = *(const float4*)(row + kbase + 4);
  float4 f2 = *(const float4*)(row + kbase + 16);
  float4 f3 = *(const float4*)(row + kbase + 20);
  v16h a;
  a[0]=(_Float16)f0.x; a[1]=(_Float16)f0.y; a[2]=(_Float16)f0.z; a[3]=(_Float16)f0.w;
  a[4]=(_Float16)f1.x; a[5]=(_Float16)f1.y; a[6]=(_Float16)f1.z; a[7]=(_Float16)f1.w;
  a[8]=(_Float16)f2.x; a[9]=(_Float16)f2.y; a[10]=(_Float16)f2.z; a[11]=(_Float16)f2.w;
  a[12]=(_Float16)f3.x; a[13]=(_Float16)f3.y; a[14]=(_Float16)f3.z; a[15]=(_Float16)f3.w;
  return a;
}

static __device__ __forceinline__ v16h make_afrag_lds(const _Float16* row, int kbase) {
  v8h lo = *(const v8h*)(row + kbase);
  v8h hi = *(const v8h*)(row + kbase + 16);
  return cat8(lo, hi);
}

// B operand (K x N = 32x16): lane n<16 = col N=n holding K=0..7,16..23; lanes 16..31
// hold K=8..15,24..31. For Y = X @ W^T, B[k][n] = W[n][k] -> lane n reads
// halves koff..koff+7 and koff+16..koff+23 of W row n.
static __device__ __forceinline__ v16h make_bfrag(const _Float16* w, int n, int koff) {
  const _Float16* p = w + n * EMBED + koff;
  v8h lo = *(const v8h*)p;
  v8h hi = *(const v8h*)(p + 16);
  return cat8(lo, hi);
}

// Stage one 32 KB f16 weight matrix from global into LDS using the CDNA5
// async DMA path (no VGPR round trip, tracked by ASYNCcnt). Each thread
// issues 8 x 16-byte copies; 256 threads cover 32 KB.
static __device__ __forceinline__ void stage_weight_async(const _Float16* __restrict__ wsrc,
                                                          _Float16* dst_lds, int tid) {
  // generic LDS pointers carry the LDS byte offset in their low 32 bits
  unsigned lds0 = (unsigned)(uintptr_t)dst_lds + (unsigned)tid * 16u;
  unsigned g0   = (unsigned)tid * 16u;
#pragma unroll
  for (int sweep = 0; sweep < 8; ++sweep) {
    asm volatile("global_load_async_to_lds_b128 %0, %1, %2"
                 :
                 : "v"(lds0 + sweep * 4096u), "v"(g0 + sweep * 4096u), "s"(wsrc)
                 : "memory");
  }
}

static __device__ __forceinline__ void wait_async0() {
  asm volatile("s_wait_asynccnt 0x0" ::: "memory");
}

// Prep: convert the four 128x128 f32 weights to f16 once into d_ws.
__global__ void convert_weights_kernel(const float* __restrict__ Wq, const float* __restrict__ Wk,
                                       const float* __restrict__ Wv, const float* __restrict__ Wo,
                                       _Float16* __restrict__ w16) {
  int i = (blockIdx.x * blockDim.x + threadIdx.x) * 4;   // 4096 threads * 4 = 16384
  const float* src[4] = {Wq, Wk, Wv, Wo};
#pragma unroll
  for (int m = 0; m < 4; ++m) {
    float4 f = *(const float4*)(src[m] + i);
    v4h h = { (_Float16)f.x, (_Float16)f.y, (_Float16)f.z, (_Float16)f.w };
    *(v4h*)(w16 + m * WELEMS + i) = h;
  }
}

__global__ __launch_bounds__(THREADS, 1)
void mha_fused_kernel(const float* __restrict__ x,
                      const _Float16* __restrict__ w16,   // [4][128*128] f16: q,k,v,o
                      const float* __restrict__ bo,
                      float* __restrict__ out)
{
  __shared__ Smem sm;
  const int tid  = threadIdx.x;
  const int lane = tid & 31;
  const int wave = tid >> 5;
  const int l15  = lane & 15;
  const int hi16 = lane >> 4;           // 0: lanes 0-15, 1: lanes 16-31

  for (int tile = blockIdx.x; tile < NTILES; tile += gridDim.x) {
    // L2 prefetch of the next tile's activations (global_prefetch_b8)
    {
      int nxt = tile + gridDim.x;
      if (nxt < NTILES) {
        const char* p = (const char*)(x + (size_t)nxt * TOK_PER_BLOCK * EMBED) + tid * 256;
        __builtin_prefetch(p, 0, 0);
        __builtin_prefetch(p + 128, 0, 0);
      }
    }

    // Kick off the async DMA of Wq into LDS, then hide it behind the long
    // global f32 loads of this wave's 16 x-rows (kept as A fragments in VGPRs
    // and reused across all three projections).
    stage_weight_async(w16 + 0 * WELEMS, sm.wstage, tid);

    const int trow = wave * 16 + l15;
    const float* xrow = x + ((size_t)tile * TOK_PER_BLOCK + trow) * EMBED;
    v16h afrag[4];
#pragma unroll
    for (int ks = 0; ks < 4; ++ks)
      afrag[ks] = make_afrag_f32(xrow, ks * 32 + hi16 * 8);

    // One projection: Y = A @ W^T via 8 N-tiles x 4 K-steps of WMMA.
    // B fragments are preloaded per N-tile so the 4 WMMAs issue back-to-back.
    auto project = [&](int proj) {
#pragma unroll
      for (int nt = 0; nt < 8; ++nt) {
        v16h b[4];
#pragma unroll
        for (int ks = 0; ks < 4; ++ks)
          b[ks] = make_bfrag(sm.wstage, nt * 16 + l15, ks * 32 + hi16 * 8);
        v8f acc = {};
#pragma unroll
        for (int ks = 0; ks < 4; ++ks)
          acc = __builtin_amdgcn_wmma_f32_16x16x32_f16(false, afrag[ks], false, b[ks],
                                                       (short)0, acc, false, false);
        const int col = nt * 16 + l15;
#pragma unroll
        for (int r = 0; r < 8; ++r)   // C layout: VGPR r -> M = r (+8 for hi lanes)
          sm.qkv[proj][(wave * 16 + r + hi16 * 8) * EMBED + col] = (_Float16)acc[r];
      }
    };

    wait_async0();        // Wq resident
    __syncthreads();
    project(0);
    __syncthreads();

    stage_weight_async(w16 + 1 * WELEMS, sm.wstage, tid);
    wait_async0();
    __syncthreads();
    project(1);
    __syncthreads();

    stage_weight_async(w16 + 2 * WELEMS, sm.wstage, tid);
    wait_async0();
    __syncthreads();
    project(2);
    __syncthreads();

    // DMA Wo while the attention phase runs (it only touches qkv)
    stage_weight_async(w16 + 3 * WELEMS, sm.wstage, tid);

    // ---- per-token attention over the heads axis; lane = query head.
    // scores = 0.5 * Q K^T (32x32), softmax over key heads, O = A V.
    // Entirely wave-local (each wave owns its 16 token rows of qkv).
#pragma unroll 1
    for (int ti = 0; ti < 16; ++ti) {
      const int row = wave * 16 + ti;
      const _Float16* qrow = &sm.qkv[0][row * EMBED];
      const _Float16* krow = &sm.qkv[1][row * EMBED];
      const _Float16* vrow = &sm.qkv[2][row * EMBED];
      v4h qh = *(const v4h*)(qrow + lane * HDIM);
      float q0 = (float)qh[0], q1 = (float)qh[1], q2 = (float)qh[2], q3 = (float)qh[3];
      float s[HEADS];
#pragma unroll
      for (int kh = 0; kh < HEADS; kh += 2) {   // 2 key heads per ds_load_b128
        v8h kk = *(const v8h*)(krow + kh * HDIM);
        s[kh]     = 0.5f * (q0*(float)kk[0] + q1*(float)kk[1] + q2*(float)kk[2] + q3*(float)kk[3]);
        s[kh + 1] = 0.5f * (q0*(float)kk[4] + q1*(float)kk[5] + q2*(float)kk[6] + q3*(float)kk[7]);
      }
      float mx = s[0];
#pragma unroll
      for (int kh = 1; kh < HEADS; ++kh) mx = fmaxf(mx, s[kh]);
      float sum = 0.f;
#pragma unroll
      for (int kh = 0; kh < HEADS; ++kh) { s[kh] = __expf(s[kh] - mx); sum += s[kh]; }
      float inv = 1.f / sum;
      float o0 = 0.f, o1 = 0.f, o2 = 0.f, o3 = 0.f;
#pragma unroll
      for (int kh = 0; kh < HEADS; kh += 2) {
        v8h vv = *(const v8h*)(vrow + kh * HDIM);
        o0 += s[kh]*(float)vv[0] + s[kh+1]*(float)vv[4];
        o1 += s[kh]*(float)vv[1] + s[kh+1]*(float)vv[5];
        o2 += s[kh]*(float)vv[2] + s[kh+1]*(float)vv[6];
        o3 += s[kh]*(float)vv[3] + s[kh+1]*(float)vv[7];
      }
      v4h ov = { (_Float16)(o0 * inv), (_Float16)(o1 * inv),
                 (_Float16)(o2 * inv), (_Float16)(o3 * inv) };
      // attended row overlays Q (Q[row] fully consumed above)
      *(v4h*)(&sm.qkv[0][row * EMBED + lane * HDIM]) = ov;
    }

    wait_async0();        // Wo resident
    __syncthreads();

    // ---- output projection: out = attended @ Wo^T + bo
    v16h ofrag[4];
    const _Float16* arow = &sm.qkv[0][trow * EMBED];
#pragma unroll
    for (int ks = 0; ks < 4; ++ks)
      ofrag[ks] = make_afrag_lds(arow, ks * 32 + hi16 * 8);

    float* orow = out + (size_t)tile * TOK_PER_BLOCK * EMBED;
#pragma unroll
    for (int nt = 0; nt < 8; ++nt) {
      v16h b[4];
#pragma unroll
      for (int ks = 0; ks < 4; ++ks)
        b[ks] = make_bfrag(sm.wstage, nt * 16 + l15, ks * 32 + hi16 * 8);
      v8f acc = {};
#pragma unroll
      for (int ks = 0; ks < 4; ++ks)
        acc = __builtin_amdgcn_wmma_f32_16x16x32_f16(false, ofrag[ks], false, b[ks],
                                                     (short)0, acc, false, false);
      const int col = nt * 16 + l15;
      float bias = bo[col];
#pragma unroll
      for (int r = 0; r < 8; ++r)
        orow[(wave * 16 + r + hi16 * 8) * EMBED + col] = acc[r] + bias;
    }
    __syncthreads();   // wstage is restaged next iteration
  }
}

extern "C" void kernel_launch(void* const* d_in, const int* in_sizes, int n_in,
                              void* d_out, int out_size, void* d_ws, size_t ws_size,
                              hipStream_t stream) {
  (void)in_sizes; (void)n_in; (void)out_size; (void)ws_size;
  const float* x  = (const float*)d_in[0];
  const float* Wq = (const float*)d_in[1];
  const float* Wk = (const float*)d_in[2];
  const float* Wv = (const float*)d_in[3];
  const float* Wo = (const float*)d_in[4];
  const float* bo = (const float*)d_in[5];
  float* out = (float*)d_out;
  _Float16* w16 = (_Float16*)d_ws;   // 4 * 16384 f16 = 128 KB scratch

  hipLaunchKernelGGL(convert_weights_kernel, dim3(16), dim3(256), 0, stream,
                     Wq, Wk, Wv, Wo, w16);
  hipLaunchKernelGGL(mha_fused_kernel, dim3(NTILES), dim3(THREADS), 0, stream,
                     x, w16, bo, out);
}
